// TextConditionedDynamicLayerAttention_18769007084493
// MI455X (gfx1250) — matrix operand order repaired and checked by reading, not compiled, not run on hardware
//
#include <hip/hip_runtime.h>
#include <hip/hip_bf16.h>
#include <math.h>

// ---------------------------------------------------------------------------
// Problem constants (from reference): D=4096, L=24, H=8, HD=512, R=1024,
// B=1, T=128, N=576, S=L*N=13824.
// ---------------------------------------------------------------------------
#define DD  4096
#define LL  24
#define HH  8
#define HDD 512
#define RR  1024
#define TT  128
#define NNV 576
#define SSV (LL*NNV)   // 13824

typedef __attribute__((ext_vector_type(16))) __bf16 v16bf;
typedef __attribute__((ext_vector_type(8)))  float  v8f;

__device__ __forceinline__ float sigf(float x){ return 1.f/(1.f+expf(-x)); }

// ---------------------------------------------------------------------------
// Generic WMMA tile GEMM:
//   Out[m,n] = act( scale * sum_k (A[m,k] + addv[(m/rowdiv),k]) * W(n,k) + bias[n] ) * emul[m,n]
// W(n,k) = Wm[n*ldw + k]  (TRANSW=false)  or  Wm[k*ldw + n]  (TRANSW=true)
// Batched over blockIdx.z with element strides Ahs/Whs/Ohs/Ehs.
// BM=128 x BN=128, BK=32, 256 threads (8 wave32): each wave owns a 16-row
// strip and eight 16x16 v_wmma_f32_16x16x32_bf16 accumulators (A-frag reused
// across all 8 B-frags). bf16-resident operands are staged into LDS with
// GLOBAL_LOAD_ASYNC_TO_LDS_B128 (ASYNCcnt); f32 operands go through VGPRs
// with v_cvt_pk_bf16_f32 since they need conversion anyway.
// ---------------------------------------------------------------------------
constexpr int BM = 128, BN = 128, BK = 32;
constexpr int LPAD = 8;   // row stride (BK+LPAD)*2 = 80B: 16B-aligned, 20-bank stride

template<typename AT, typename WT, typename OT, int ACT, bool TRANSW>
__global__ __launch_bounds__(256) void gemm_kernel(
    const AT* __restrict__ A, const WT* __restrict__ Wm,
    const float* __restrict__ bias, const float* __restrict__ addv,
    const float* __restrict__ emul, OT* __restrict__ Out,
    int M, int N, int K, int lda, int ldw, int ldo, int lde,
    long Ahs, long Whs, long Ohs, long Ehs, int rowdiv, float scale)
{
  __shared__ __bf16 As[BM][BK + LPAD];
  __shared__ __bf16 Bs[BN][BK + LPAD];

  const int tid = threadIdx.x;
  const int bm0 = blockIdx.y * BM;
  const int bn0 = blockIdx.x * BN;
  const int z   = blockIdx.z;
  A   += (long)z * Ahs;
  Wm  += (long)z * Whs;
  Out += (long)z * Ohs;
  const float* em = emul ? (emul + (long)z * Ehs) : nullptr;

  const int lane = tid & 31, wv = tid >> 5;
  const int lh = lane >> 4, lm = lane & 15;

  v8f acc[8];
  #pragma unroll
  for (int t = 0; t < 8; t++)
    #pragma unroll
    for (int i = 0; i < 8; i++) acc[t][i] = 0.f;

  // cooperative-load assignment: each thread stages 16 elements (32B) of A
  // and 16 elements of W per k-step.
  const int ar = tid >> 1;            // tile row 0..127
  const int ak = (tid & 1) * 16;      // k offset 0 or 16
  int agm = bm0 + ar; if (agm >= M) agm = M - 1;
  int wgn = bn0 + ar; if (wgn >= N) wgn = N - 1;

  for (int k0 = 0; k0 < K; k0 += BK) {
    // ---- stage A tile ----
    if constexpr (sizeof(AT) == 2) {
      // bf16 in global: async DMA straight into LDS (GV mode, ASYNCcnt)
      unsigned ldsa = (unsigned)(unsigned long long)&As[ar][ak];
      unsigned long long ga = (unsigned long long)(A + (long)agm * lda + k0 + ak);
      asm volatile(
        "global_load_async_to_lds_b128 %0, %1, off\n\t"
        "global_load_async_to_lds_b128 %0, %1, off offset:16"
        :: "v"(ldsa), "v"(ga) : "memory");
    } else {
      const AT* ap = A + (long)agm * lda + k0 + ak;
      float v[16];
      #pragma unroll
      for (int j = 0; j < 16; j++) v[j] = (float)ap[j];
      if (addv) {
        const float* cv = addv + (long)(agm / rowdiv) * K + k0 + ak;
        #pragma unroll
        for (int j = 0; j < 16; j++) v[j] += cv[j];
      }
      #pragma unroll
      for (int j = 0; j < 16; j++) As[ar][ak + j] = (__bf16)v[j];
      __builtin_prefetch(ap + BK, 0, 0);   // global_prefetch_b8 (next k-tile)
    }
    // ---- stage W tile ----
    if constexpr (!TRANSW && sizeof(WT) == 2) {
      unsigned ldsb = (unsigned)(unsigned long long)&Bs[ar][ak];
      unsigned long long gw = (unsigned long long)(Wm + (long)wgn * ldw + k0 + ak);
      asm volatile(
        "global_load_async_to_lds_b128 %0, %1, off\n\t"
        "global_load_async_to_lds_b128 %0, %1, off offset:16"
        :: "v"(ldsb), "v"(gw) : "memory");
    } else if constexpr (!TRANSW) {
      const WT* wp = Wm + (long)wgn * ldw + k0 + ak;
      #pragma unroll
      for (int j = 0; j < 16; j++) Bs[ar][ak + j] = (__bf16)(float)wp[j];
      __builtin_prefetch(wp + BK, 0, 0);
    } else {
      #pragma unroll
      for (int j = 0; j < 16; j++)
        Bs[ar][ak + j] = (__bf16)(float)Wm[(long)(k0 + ak + j) * ldw + wgn];
    }
    if constexpr (sizeof(AT) == 2 || (!TRANSW && sizeof(WT) == 2)) {
      asm volatile("s_wait_asynccnt 0" ::: "memory");
    }
    __syncthreads();

    // A fragment: lane = M row (lane&15); half-wave picks K 16-group
    v16bf af;
    const int am = wv * 16 + lm;
    #pragma unroll
    for (int j = 0; j < 16; j++) {
      const int kk = lh * 16 + j;
      af[j] = As[am][kk];
    }
    // B fragments: lane = N column; half-wave picks K 16-group
    #pragma unroll
    for (int t = 0; t < 8; t++) {
      v16bf bfr;
      const int bn = t * 16 + lm;
      const int kb = lh * 16;
      #pragma unroll
      for (int j = 0; j < 16; j++) bfr[j] = Bs[bn][kb + j];
      acc[t] = __builtin_amdgcn_wmma_f32_16x16x32_bf16(
          false, af, false, bfr, (short)0, acc[t], false, false);
    }
    __syncthreads();
  }

  // epilogue: C layout VGPR i -> M = i + 8*(lane>=16), N = lane&15
  #pragma unroll
  for (int t = 0; t < 8; t++) {
    #pragma unroll
    for (int i = 0; i < 8; i++) {
      const int gm = bm0 + wv * 16 + lh * 8 + i;
      const int gn = bn0 + t * 16 + lm;
      if (gm < M && gn < N) {
        float v = acc[t][i] * scale;
        if (bias) v += bias[gn];
        if (ACT == 1) v = 0.5f * v * (1.f + erff(v * 0.70710678118f)); // exact GELU
        else if (ACT == 2) v = sigf(v);
        if (em) v *= em[(long)gm * lde + gn];
        Out[(long)gm * ldo + gn] = (OT)v;
      }
    }
  }
}

// ---------------------------------------------------------------------------
// Mean over `rows` rows of a [rows, DD] group; groups along blockIdx.y.
// ---------------------------------------------------------------------------
__global__ __launch_bounds__(256) void mean_rows_kernel(
    const float* __restrict__ in, float* __restrict__ out, int rows)
{
  const int d = blockIdx.x * 256 + threadIdx.x;
  const long g = blockIdx.y;
  const float* p = in + g * (long)rows * DD + d;
  float s = 0.f;
  for (int r = 0; r < rows; r++) s += p[(long)r * DD];
  out[g * DD + d] = s / (float)rows;
}

// ---------------------------------------------------------------------------
// DSU recurrence, step 1: s = relu(W1 @ [sigmoid(c_prev), y_l, tg] + W1_b)
// One wave per output (R=1024 outputs). cprev==nullptr -> sigmoid(0)=0.5.
// ---------------------------------------------------------------------------
__global__ __launch_bounds__(256) void dsu_s_kernel(
    const float* __restrict__ W1, const float* __restrict__ W1b,
    const float* __restrict__ cprev, const float* __restrict__ yl,
    const float* __restrict__ tg, float* __restrict__ s)
{
  const int r = blockIdx.x * 8 + (threadIdx.x >> 5);
  const int lane = threadIdx.x & 31;
  const float* wr = W1 + (long)r * (3 * DD);
  float acc = 0.f;
  for (int k = lane; k < 3 * DD; k += 32) {
    float x;
    if (k < DD)            x = cprev ? sigf(cprev[k]) : 0.5f;
    else if (k < 2 * DD)   x = yl[k - DD];
    else                   x = tg[k - 2 * DD];
    acc += wr[k] * x;
  }
  #pragma unroll
  for (int o = 16; o; o >>= 1) acc += __shfl_xor(acc, o, 32);
  if (lane == 0) s[r] = fmaxf(acc + W1b[r], 0.f);
}

// ---------------------------------------------------------------------------
// DSU recurrence, step 2: c = sigmoid(f)*c_prev + sigmoid(i)*tanh(ct)
// One wave per output d (D=4096 outputs); three dot products in one sweep.
// ---------------------------------------------------------------------------
__global__ __launch_bounds__(256) void dsu_c_kernel(
    const float* __restrict__ Wc, const float* __restrict__ Wcb,
    const float* __restrict__ Wi, const float* __restrict__ Wib,
    const float* __restrict__ Wf, const float* __restrict__ Wfb,
    const float* __restrict__ bc, const float* __restrict__ bi,
    const float* __restrict__ bfb,
    const float* __restrict__ s, const float* __restrict__ cprev,
    float* __restrict__ cout)
{
  const int d = blockIdx.x * 8 + (threadIdx.x >> 5);
  const int lane = threadIdx.x & 31;
  const float* wc = Wc + (long)d * RR;
  const float* wi = Wi + (long)d * RR;
  const float* wf = Wf + (long)d * RR;
  float ac = 0.f, ai = 0.f, af = 0.f;
  for (int k = lane; k < RR; k += 32) {
    const float sv = s[k];
    ac += wc[k] * sv; ai += wi[k] * sv; af += wf[k] * sv;
  }
  #pragma unroll
  for (int o = 16; o; o >>= 1) {
    ac += __shfl_xor(ac, o, 32);
    ai += __shfl_xor(ai, o, 32);
    af += __shfl_xor(af, o, 32);
  }
  if (lane == 0) {
    const float ct = tanhf(ac + Wcb[d] + bc[d]);
    const float ig = sigf(ai + Wib[d] + bi[d]);
    const float fg = sigf(af + Wfb[d] + bfb[d]);
    const float cp = cprev ? cprev[d] : 0.f;
    cout[d] = fg * cp + ig * ct;
  }
}

// ---------------------------------------------------------------------------
// Row LayerNorm (population var, eps=1e-5); may run in-place (per-element
// ownership: each thread re-reads only its own elements before writing them).
// ---------------------------------------------------------------------------
template<typename IT, typename OT>
__global__ __launch_bounds__(256) void ln_kernel(
    const IT* __restrict__ in, OT* __restrict__ out,
    const float* __restrict__ g, const float* __restrict__ b,
    int cols, long ldi, long ldo)
{
  __shared__ float shs[8], shs2[8];
  const long row = blockIdx.x;
  const IT* x = in + row * ldi;
  OT* o = out + row * ldo;
  const int tid = threadIdx.x;
  float s = 0.f, s2 = 0.f;
  for (int c = tid; c < cols; c += 256) {
    const float v = (float)x[c]; s += v; s2 += v * v;
  }
  #pragma unroll
  for (int off = 16; off; off >>= 1) {
    s += __shfl_xor(s, off, 32); s2 += __shfl_xor(s2, off, 32);
  }
  if ((tid & 31) == 0) { shs[tid >> 5] = s; shs2[tid >> 5] = s2; }
  __syncthreads();
  if (tid == 0) {
    float a = 0.f, a2 = 0.f;
    for (int w = 0; w < 8; w++) { a += shs[w]; a2 += shs2[w]; }
    shs[0] = a; shs2[0] = a2;
  }
  __syncthreads();
  const float mean = shs[0] / (float)cols;
  const float var  = shs2[0] / (float)cols - mean * mean;
  const float inv  = rsqrtf(var + 1e-5f);
  for (int c = tid; c < cols; c += 256)
    o[c] = (OT)(((float)x[c] - mean) * inv * g[c] + b[c]);
}

// ---------------------------------------------------------------------------
// In-place row softmax over length S (scores already scaled by GEMM epilogue).
// ---------------------------------------------------------------------------
__global__ __launch_bounds__(256) void softmax_kernel(float* __restrict__ sc, int S)
{
  __shared__ float sh[8];
  const long row = blockIdx.x;
  float* x = sc + row * (long)S;
  const int tid = threadIdx.x;

  float mx = -3.4e38f;
  for (int c = tid; c < S; c += 256) mx = fmaxf(mx, x[c]);
  #pragma unroll
  for (int off = 16; off; off >>= 1) mx = fmaxf(mx, __shfl_xor(mx, off, 32));
  if ((tid & 31) == 0) sh[tid >> 5] = mx;
  __syncthreads();
  if (tid == 0) { float m = sh[0]; for (int w = 1; w < 8; w++) m = fmaxf(m, sh[w]); sh[0] = m; }
  __syncthreads();
  mx = sh[0];
  __syncthreads();

  float s = 0.f;
  for (int c = tid; c < S; c += 256) { const float e = expf(x[c] - mx); x[c] = e; s += e; }
  #pragma unroll
  for (int off = 16; off; off >>= 1) s += __shfl_xor(s, off, 32);
  if ((tid & 31) == 0) sh[tid >> 5] = s;
  __syncthreads();
  if (tid == 0) { float a = 0.f; for (int w = 0; w < 8; w++) a += sh[w]; sh[0] = a; }
  __syncthreads();
  const float inv = 1.f / sh[0];
  for (int c = tid; c < S; c += 256) x[c] *= inv;
}

// ---------------------------------------------------------------------------
// Host-side orchestration
// ---------------------------------------------------------------------------
extern "C" void kernel_launch(void* const* d_in, const int* in_sizes, int n_in,
                              void* d_out, int out_size, void* d_ws, size_t ws_size,
                              hipStream_t stream)
{
  (void)in_sizes; (void)n_in; (void)out_size; (void)ws_size;
  const float* text = (const float*)d_in[0];
  const float* feats= (const float*)d_in[1];
  const float* W1_w = (const float*)d_in[2];
  const float* W1_b = (const float*)d_in[3];
  const float* Wc_w = (const float*)d_in[4];
  const float* Wc_b = (const float*)d_in[5];
  const float* Wi_w = (const float*)d_in[6];
  const float* Wi_b = (const float*)d_in[7];
  const float* Wf_w = (const float*)d_in[8];
  const float* Wf_b = (const float*)d_in[9];
  const float* bc   = (const float*)d_in[10];
  const float* bi   = (const float*)d_in[11];
  const float* bf_  = (const float*)d_in[12];
  const float* g1_w = (const float*)d_in[13];
  const float* g1_b = (const float*)d_in[14];
  const float* g2_w = (const float*)d_in[15];
  const float* g2_b = (const float*)d_in[16];
  const float* Wq_w = (const float*)d_in[17];
  const float* Wq_b = (const float*)d_in[18];
  const float* Wk_w = (const float*)d_in[19];
  const float* Wk_b = (const float*)d_in[20];
  const float* Wo_w = (const float*)d_in[21];
  const float* Wo_b = (const float*)d_in[22];
  const float* qn_g = (const float*)d_in[23];
  const float* qn_b = (const float*)d_in[24];
  const float* kn_g = (const float*)d_in[25];
  const float* kn_b = (const float*)d_in[26];
  const float* on_g = (const float*)d_in[27];
  const float* on_b = (const float*)d_in[28];

  char* ws = (char*)d_ws;
  size_t off = 0;
  auto alloc = [&](size_t bytes) -> char* {
    char* p = ws + off;
    off = (off + bytes + 255) & ~(size_t)255;
    return p;
  };
  float*  tg     = (float*) alloc((size_t)DD * 4);
  float*  ybuf   = (float*) alloc((size_t)LL * DD * 4);
  float*  cbuf   = (float*) alloc((size_t)LL * DD * 4);
  float*  sbuf   = (float*) alloc((size_t)RR * 4);
  __bf16* Hbuf   = (__bf16*)alloc((size_t)SSV * DD * 2);   // gelu(x@g1^T)
  __bf16* Rbuf   = (__bf16*)alloc((size_t)SSV * DD * 2);   // refreshed (also V)
  __bf16* Kbuf   = (__bf16*)alloc((size_t)SSV * DD * 2);   // K (LN in place)
  float*  Qpre   = (float*) alloc((size_t)TT * DD * 4);
  __bf16* Qbuf   = (__bf16*)alloc((size_t)TT * DD * 2);
  float*  scores = (float*) alloc((size_t)HH * TT * SSV * 4);
  float*  attb   = (float*) alloc((size_t)TT * DD * 4);
  float*  obuf   = (float*) alloc((size_t)TT * DD * 4);

  // 1) means: text_global, per-layer y
  mean_rows_kernel<<<dim3(DD/256, 1),  256, 0, stream>>>(text,  tg,   TT);
  mean_rows_kernel<<<dim3(DD/256, LL), 256, 0, stream>>>(feats, ybuf, NNV);

  // 2) DSU recurrence (sequential over L layers; weights L2-resident)
  for (int l = 0; l < LL; l++) {
    const float* cprev = l ? (const float*)(cbuf + (size_t)(l-1)*DD) : (const float*)nullptr;
    dsu_s_kernel<<<RR/8, 256, 0, stream>>>(W1_w, W1_b, cprev, ybuf + (size_t)l*DD, tg, sbuf);
    dsu_c_kernel<<<DD/8, 256, 0, stream>>>(Wc_w, Wc_b, Wi_w, Wi_b, Wf_w, Wf_b,
                                           bc, bi, bf_, sbuf, cprev, cbuf + (size_t)l*DD);
  }

  // 3) spatial gate GEMM 1: H = gelu((feats + c[l]) @ g1^T + g1_b)   [S, D]
  gemm_kernel<float, float, __bf16, 1, false>
    <<<dim3(DD/BN, SSV/BM, 1), 256, 0, stream>>>(
      feats, g1_w, g1_b, cbuf, nullptr, Hbuf,
      SSV, DD, DD, DD, DD, DD, 0, 0, 0, 0, 0, NNV, 1.f);

  // 4) gate GEMM 2: refreshed = feats * sigmoid(H @ g2^T + g2_b)
  gemm_kernel<__bf16, float, __bf16, 2, false>
    <<<dim3(DD/BN, SSV/BM, 1), 256, 0, stream>>>(
      Hbuf, g2_w, g2_b, nullptr, feats, Rbuf,
      SSV, DD, DD, DD, DD, DD, DD, 0, 0, 0, 0, 1, 1.f);

  // 5) K projection + row-LN (in place, bf16)
  gemm_kernel<__bf16, float, __bf16, 0, false>
    <<<dim3(DD/BN, SSV/BM, 1), 256, 0, stream>>>(
      Rbuf, Wk_w, Wk_b, nullptr, nullptr, Kbuf,
      SSV, DD, DD, DD, DD, DD, 0, 0, 0, 0, 0, 1, 1.f);
  ln_kernel<__bf16, __bf16><<<SSV, 256, 0, stream>>>(Kbuf, Kbuf, kn_g, kn_b, DD, DD, DD);

  // 6) Q projection + LN -> bf16
  gemm_kernel<float, float, float, 0, false>
    <<<dim3(DD/BN, TT/BM, 1), 256, 0, stream>>>(
      text, Wq_w, Wq_b, nullptr, nullptr, Qpre,
      TT, DD, DD, DD, DD, DD, 0, 0, 0, 0, 0, 1, 1.f);
  ln_kernel<float, __bf16><<<TT, 256, 0, stream>>>(Qpre, Qbuf, qn_g, qn_b, DD, DD, DD);

  // 7) scores = Q @ K^T / sqrt(HD*L), batched over heads (z); A and W bf16 ->
  //    both operands staged with global_load_async_to_lds_b128.
  const float scl = 1.f / sqrtf((float)HDD * (float)LL);
  gemm_kernel<__bf16, __bf16, float, 0, false>
    <<<dim3(SSV/BN, TT/BM, HH), 256, 0, stream>>>(
      Qbuf, Kbuf, nullptr, nullptr, nullptr, scores,
      TT, SSV, HDD, DD, DD, SSV, 0,
      (long)HDD, (long)HDD, (long)TT*SSV, 0, 1, scl);

  // 8) softmax over S, in place
  softmax_kernel<<<HH*TT, 256, 0, stream>>>(scores, SSV);

  // 9) att = P @ V  (V = Rbuf, accessed transposed: W(n=d,k=s)=Rbuf[s*D + h*HD + d])
  gemm_kernel<float, __bf16, float, 0, true>
    <<<dim3(HDD/BN, TT/BM, HH), 256, 0, stream>>>(
      scores, Rbuf, nullptr, nullptr, nullptr, attb,
      TT, HDD, SSV, SSV, DD, DD, 0,
      (long)TT*SSV, (long)HDD, (long)HDD, 0, 1, 1.f);

  // 10) output projection + LN -> d_out (f32)
  gemm_kernel<float, float, float, 0, false>
    <<<dim3(DD/BN, TT/BM, 1), 256, 0, stream>>>(
      attb, Wo_w, Wo_b, nullptr, nullptr, obuf,
      TT, DD, DD, DD, DD, DD, 0, 0, 0, 0, 0, 1, 1.f);
  ln_kernel<float, float><<<TT, 256, 0, stream>>>(obuf, (float*)d_out, on_g, on_b, DD, DD, DD);
}
